// GCNSynthetic_22127671509522
// MI455X (gfx1250) — matrix operand order repaired and checked by reading, not compiled, not run on hardware
//
#include <hip/hip_runtime.h>
#include <cstdint>

#define NROWS 16384
#define HID   64
#define NCLS  10

typedef __attribute__((ext_vector_type(16))) __bf16 v16bf;
typedef __attribute__((ext_vector_type(8)))  float  v8f;

union BFrag {
    v16bf v;
    uint4 q[2];
    unsigned int u[8];
};

struct RawA { float4 x0, x1, x2, x3; };
struct RawB { uint4 q0, q1; };

__device__ __forceinline__ RawA loadA(const float* __restrict__ p) {
    RawA r;
    r.x0 = *(const float4*)(p);
    r.x1 = *(const float4*)(p + 4);
    r.x2 = *(const float4*)(p + 16);
    r.x3 = *(const float4*)(p + 20);
    return r;
}

__device__ __forceinline__ RawB loadB(const unsigned short* __restrict__ p) {
    RawB r;
    r.q0 = *(const uint4*)(p);
    r.q1 = *(const uint4*)(p + 16);
    return r;
}

// two f32 -> packed bf16 pair: round-half-up then grab high halves with v_perm_b32
__device__ __forceinline__ unsigned int pk_bf16(float a, float b) {
    unsigned int ua = __builtin_bit_cast(unsigned int, a) + 0x8000u;
    unsigned int ub = __builtin_bit_cast(unsigned int, b) + 0x8000u;
    // dest bytes = {ua.b2, ua.b3, ub.b2, ub.b3} ; src0=ub (bytes 4-7), src1=ua (bytes 0-3)
    return __builtin_amdgcn_perm(ub, ua, 0x07060302u);
}

__device__ __forceinline__ void cvtA(const RawA& r, BFrag& a) {
    a.u[0] = pk_bf16(r.x0.x, r.x0.y);  a.u[1] = pk_bf16(r.x0.z, r.x0.w);
    a.u[2] = pk_bf16(r.x1.x, r.x1.y);  a.u[3] = pk_bf16(r.x1.z, r.x1.w);
    a.u[4] = pk_bf16(r.x2.x, r.x2.y);  a.u[5] = pk_bf16(r.x2.z, r.x2.w);
    a.u[6] = pk_bf16(r.x3.x, r.x3.y);  a.u[7] = pk_bf16(r.x3.z, r.x3.w);
}

// ---------------------------------------------------------------------------
// feat: support = h @ W  -> stored transposed bf16:  ST[col * NROWS + row]
// ---------------------------------------------------------------------------
__global__ __launch_bounds__(256) void feat_kernel(
    const float* __restrict__ h, const float* __restrict__ W,
    int in_dim, unsigned short* __restrict__ ST) {
    int id  = blockIdx.x * 256 + threadIdx.x;   // NROWS*64 threads
    int row = id >> 6;
    int col = id & 63;
    const float* hr = h + (long)row * in_dim;
    float s = 0.0f;
    for (int k = 0; k < in_dim; ++k)
        s += hr[k] * W[k * HID + col];          // hr[k] broadcast, W coalesced
    unsigned int u = __builtin_bit_cast(unsigned int, s);
    u = (u + 0x7FFFu + ((u >> 16) & 1u)) >> 16;
    ST[(long)col * NROWS + row] = (unsigned short)u;
}

// ---------------------------------------------------------------------------
// agg: hout = relu(adj @ S + bias)
// Block = 128 threads = 4 waves; wave -> 16 rows x 64 cols; grid = NROWS/64.
// K loop step 32, v_wmma_f32_16x16x32_bf16, software-pipelined one K-step
// ahead so WMMAs overlap the next iteration's VMEM.
//
// A frag (16x32 bf16): lane l<16 -> row l, K {0..7, 16..23}; lanes 16..31 ->
// same rows, K {8..15, 24..31}.
// B frag (32x16 bf16): lane n<16 -> col n, K {0..7, 16..23}; lanes 16..31 ->
// K {8..15, 24..31} (two 16B chunks in the K-major ST stream).
// C frag (16x16 f32): VGPR v -> row v + 8*(lane>=16), col = lane&15.
// ---------------------------------------------------------------------------
__global__ __launch_bounds__(128) void agg_kernel(
    const float* __restrict__ adj, const unsigned short* __restrict__ ST,
    const float* __restrict__ bias, float* __restrict__ hout) {

    const int lane    = threadIdx.x & 31;
    const int wave    = threadIdx.x >> 5;            // 0..3
    const int rowBase = blockIdx.x * 64 + wave * 16;
    const int half    = lane >> 4;                   // 0 or 1
    const int l15     = lane & 15;

    const float* __restrict__ ap =
        adj + (long)(rowBase + l15) * NROWS + half * 8;
    const unsigned short* __restrict__ bp =
        ST + (long)l15 * NROWS + half * 8;

    v8f acc0 = {}, acc1 = {}, acc2 = {}, acc3 = {};

    // ---- pipeline prologue: loads for k = 0 ----
    RawA ra  = loadA(ap);
    RawB rb0 = loadB(bp);
    RawB rb1 = loadB(bp + 16L * NROWS);
    RawB rb2 = loadB(bp + 32L * NROWS);
    RawB rb3 = loadB(bp + 48L * NROWS);

    for (int k = 0; k < NROWS; k += 32) {
        // ---- issue next iteration's loads (wraps harmlessly on last iter) ----
        const int kn = (k + 32) & (NROWS - 1);
        RawA ran  = loadA(ap + kn);
        RawB rbn0 = loadB(bp + kn);
        RawB rbn1 = loadB(bp + 16L * NROWS + kn);
        RawB rbn2 = loadB(bp + 32L * NROWS + kn);
        RawB rbn3 = loadB(bp + 48L * NROWS + kn);
        __builtin_prefetch(ap + kn + 1024, 0, 1);    // global_prefetch_b8, ~4KB ahead

        // ---- compute with current data ----
        BFrag a;  cvtA(ra, a);
        {
            BFrag b; b.q[0] = rb0.q0; b.q[1] = rb0.q1;
            acc0 = __builtin_amdgcn_wmma_f32_16x16x32_bf16(
                false, a.v, false, b.v, (short)0, acc0, false, false);
        }
        {
            BFrag b; b.q[0] = rb1.q0; b.q[1] = rb1.q1;
            acc1 = __builtin_amdgcn_wmma_f32_16x16x32_bf16(
                false, a.v, false, b.v, (short)0, acc1, false, false);
        }
        {
            BFrag b; b.q[0] = rb2.q0; b.q[1] = rb2.q1;
            acc2 = __builtin_amdgcn_wmma_f32_16x16x32_bf16(
                false, a.v, false, b.v, (short)0, acc2, false, false);
        }
        {
            BFrag b; b.q[0] = rb3.q0; b.q[1] = rb3.q1;
            acc3 = __builtin_amdgcn_wmma_f32_16x16x32_bf16(
                false, a.v, false, b.v, (short)0, acc3, false, false);
        }

        // ---- rotate pipeline registers ----
        ra = ran; rb0 = rbn0; rb1 = rbn1; rb2 = rbn2; rb3 = rbn3;
    }

    // ---- bias + relu + writeback (C layout: VGPR v -> row v + 8*half) ----
    v8f accs[4] = {acc0, acc1, acc2, acc3};
    #pragma unroll
    for (int t = 0; t < 4; ++t) {
        int   c  = t * 16 + l15;
        float bb = bias[c];
        #pragma unroll
        for (int v = 0; v < 8; ++v) {
            int   m = half * 8 + v;
            float x = accs[t][v] + bb;
            hout[(long)(rowBase + m) * HID + c] = x > 0.0f ? x : 0.0f;
        }
    }
}

// ---------------------------------------------------------------------------
// final: logits = h @ Wout^T + bout ; out = log_softmax(logits)
// ---------------------------------------------------------------------------
__global__ __launch_bounds__(256) void final_kernel(
    const float* __restrict__ h, const float* __restrict__ Wout,
    const float* __restrict__ bout, float* __restrict__ out) {
    int row = blockIdx.x * 256 + threadIdx.x;
    if (row >= NROWS) return;
    const float* hr = h + (long)row * HID;
    float logits[NCLS];
    float mx = -1e30f;
    #pragma unroll
    for (int c = 0; c < NCLS; ++c) {
        float s = bout[c];
        const float* wr = Wout + c * HID;
        #pragma unroll 8
        for (int k = 0; k < HID; ++k) s += hr[k] * wr[k];
        logits[c] = s;
        mx = fmaxf(mx, s);
    }
    float sum = 0.0f;
    #pragma unroll
    for (int c = 0; c < NCLS; ++c) sum += __expf(logits[c] - mx);
    float lse = mx + __logf(sum);
    #pragma unroll
    for (int c = 0; c < NCLS; ++c) out[(long)row * NCLS + c] = logits[c] - lse;
}

// ---------------------------------------------------------------------------
extern "C" void kernel_launch(void* const* d_in, const int* in_sizes, int n_in,
                              void* d_out, int out_size, void* d_ws, size_t ws_size,
                              hipStream_t stream) {
    const float* x    = (const float*)d_in[0];
    const float* adj  = (const float*)d_in[1];
    const float* W0   = (const float*)d_in[2];
    const float* b0   = (const float*)d_in[3];
    const float* W1   = (const float*)d_in[4];
    const float* b1   = (const float*)d_in[5];
    const float* W2   = (const float*)d_in[6];
    const float* b2   = (const float*)d_in[7];
    const float* Wout = (const float*)d_in[8];
    const float* bout = (const float*)d_in[9];
    float* out = (float*)d_out;

    char* ws = (char*)d_ws;
    unsigned short* ST = (unsigned short*)ws;                  // 64*16384*2 = 2 MB
    float* hA = (float*)(ws + (2u << 20));                     // 4 MB
    float* hB = (float*)(ws + (6u << 20));                     // 4 MB

    const int featGrid = (NROWS * HID) / 256;   // 4096
    const int aggGrid  = NROWS / 64;            // 256

    // layer 0
    feat_kernel<<<featGrid, 256, 0, stream>>>(x, W0, 128, ST);
    agg_kernel<<<aggGrid, 128, 0, stream>>>(adj, ST, b0, hA);
    // layer 1
    feat_kernel<<<featGrid, 256, 0, stream>>>(hA, W1, HID, ST);
    agg_kernel<<<aggGrid, 128, 0, stream>>>(adj, ST, b1, hB);
    // layer 2
    feat_kernel<<<featGrid, 256, 0, stream>>>(hB, W2, HID, ST);
    agg_kernel<<<aggGrid, 128, 0, stream>>>(adj, ST, b2, hA);
    // output head
    final_kernel<<<NROWS / 256, 256, 0, stream>>>(hA, Wout, bout, out);
}
